// LWAN_43782896615573
// MI455X (gfx1250) — matrix operand order repaired and checked by reading, not compiled, not run on hardware
//
#include <hip/hip_runtime.h>
#include <hip/hip_bf16.h>

// ---------------------------------------------------------------------------
// LWAN fused forward for MI455X (gfx1250), wave32 + v_wmma_f32_16x16x32_bf16.
// B=8, S=512, D=768, L=4096.
// logits[b,l] = softmax_s(A·label^T)[s,l] · (H·W_out^T)[s,l]  + b_out[l]
// Kernel 2: 2(s-tiles) x 2(label-tiles) register tiling (16 b128 loads per
// 8 WMMAs) + tile-max-first online softmax (1 v_exp per score element).
// ---------------------------------------------------------------------------

typedef __attribute__((ext_vector_type(16))) __bf16 v16bf;
typedef __attribute__((ext_vector_type(8)))  float  v8f;
typedef __attribute__((ext_vector_type(4)))  int    i32x4;

union FragU { i32x4 q[2]; v16bf v; };

#define Bn 8
#define Sn 512
#define Dn 768
#define Ln 4096

__device__ __forceinline__ unsigned short f32_to_bf16(float f) {
    unsigned int u = __float_as_uint(f);
    u += 0x7FFFu + ((u >> 16) & 1u);         // round-to-nearest-even
    return (unsigned short)(u >> 16);
}

// A-operand fragment: 16x32 bf16 tile, rows row0..row0+15, cols k0..k0+31 of a
// row-major [.., ld] bf16 matrix.  ISA layout (CDNA5 ISA §7.12.2):
//   lanes 0-15  (m=lane):    VGPR0-3 = k0+0..7,   VGPR4-7 = k0+16..23
//   lanes 16-31 (m=lane-16): VGPR0-3 = k0+8..15,  VGPR4-7 = k0+24..31
__device__ __forceinline__ v16bf load_a_frag(const unsigned short* base, int ld,
                                             int row0, int k0, int lane) {
    int m = lane & 15, hi = lane >> 4;
    const unsigned short* p = base + (size_t)(row0 + m) * ld + k0 + hi * 8;
    FragU u;
    u.q[0] = *reinterpret_cast<const i32x4*>(p);        // 8 bf16
    u.q[1] = *reinterpret_cast<const i32x4*>(p + 16);   // 8 bf16, k+16
    return u.v;
}

// B-operand fragment: 32x16 bf16 where B[k,n] = rowsrc[n0+n, k0+k] (i.e. the
// 16 "columns" are 16 contiguous rows of a row-major [.., ld] matrix).
//   lanes 0-15  (n=lane):    k0+0..7,  k0+16..23
//   lanes 16-31 (n=lane-16): k0+8..15, k0+24..31
__device__ __forceinline__ v16bf load_b_frag(const unsigned short* base, int ld,
                                             int n0, int k0, int lane) {
    int n = lane & 15, hi = lane >> 4;
    const unsigned short* p = base + (size_t)(n0 + n) * ld + k0 + hi * 8;
    FragU u;
    u.q[0] = *reinterpret_cast<const i32x4*>(p);
    u.q[1] = *reinterpret_cast<const i32x4*>(p + 16);
    return u.v;
}

__device__ __forceinline__ v8f wmma_bf16(v16bf a, v16bf b, v8f c) {
    return __builtin_amdgcn_wmma_f32_16x16x32_bf16(
        /*neg_a=*/false, a, /*neg_b=*/false, b,
        /*c_mod=*/(short)0, c, /*reuse_a=*/false, /*reuse_b=*/false);
}

// ---------------------------------------------------------------------------
__global__ __launch_bounds__(256) void cvt_f32_bf16(const float* __restrict__ in,
                                                    unsigned short* __restrict__ out,
                                                    int n) {
    int i = blockIdx.x * 256 + threadIdx.x;
    if (i < n) out[i] = f32_to_bf16(in[i]);
}

// W_att is [D,D] with A_out[s,e] = sum_d H[s,d]*W_att[d,e].
// Produce WattT[e,d] = bf16(W_att[d,e]) so the B-operand reads contiguous rows.
__global__ __launch_bounds__(256) void cvt_transpose(const float* __restrict__ in,
                                                     unsigned short* __restrict__ out) {
    int i = blockIdx.x * 256 + threadIdx.x;      // writes coalesced over d
    int e = i / Dn, d = i % Dn;
    out[(size_t)e * Dn + d] = f32_to_bf16(in[(size_t)d * Dn + e]);
}

// ---------------------------------------------------------------------------
// Kernel 1: Abf = bf16(tanh(Hbf · WattT^T + b_att)), GEMM M=B*S=4096 N=768 K=768.
// One wave per 16x16 output tile; 8 waves / block.
__global__ __launch_bounds__(256) void lwan_proj_tanh(
    const unsigned short* __restrict__ Hbf,
    const unsigned short* __restrict__ WattT,
    const float* __restrict__ b_att,
    unsigned short* __restrict__ Abf) {
    int wave = blockIdx.x * 8 + (threadIdx.x >> 5);
    int lane = threadIdx.x & 31;
    const int NT = Dn / 16;                      // 48 tiles across N
    int m0 = (wave / NT) * 16;                   // row tile in flattened B*S
    int e0 = (wave % NT) * 16;                   // col tile in D

    v8f acc = {};
    for (int k0 = 0; k0 < Dn; k0 += 32) {
        v16bf a = load_a_frag(Hbf,   Dn, m0, k0, lane);
        v16bf b = load_b_frag(WattT, Dn, e0, k0, lane);
        acc = wmma_bf16(a, b, acc);
    }

    int n = lane & 15, hi = lane >> 4;
    float bias = b_att[e0 + n];
#pragma unroll
    for (int r = 0; r < 8; ++r) {
        float v = tanhf(acc[r] + bias);
        Abf[(size_t)(m0 + r + hi * 8) * Dn + e0 + n] = f32_to_bf16(v);
    }
}

// ---------------------------------------------------------------------------
// Kernel 2: fused scores + HW GEMMs + online softmax over S + logits + sigmoid.
// One wave per (batch, 32-label group); 2x2 register tiling (2 s-tiles x 2
// label-tiles): A/H fragments feed 2 label tiles, label/W_out fragments feed
// 2 s-tiles.  8 waves / block.
// C-tile layout: lane owns column n=lane&15, rows m = r + (lane>>4)*8.
__global__ __launch_bounds__(256) void lwan_attn_logits(
    const unsigned short* __restrict__ Abf,
    const unsigned short* __restrict__ Hbf,
    const unsigned short* __restrict__ Lbf,    // label_emb bf16 [L,D]
    const unsigned short* __restrict__ Wobf,   // W_out bf16 [L,D]
    const float* __restrict__ b_out,
    float* __restrict__ out) {
    int wave = blockIdx.x * 8 + (threadIdx.x >> 5);
    int lane = threadIdx.x & 31;
    int b  = wave >> 7;                        // 128 label groups per batch
    int l0 = (wave & 127) * 32;                // 32 labels per wave

    const unsigned short* Ab = Abf + (size_t)b * Sn * Dn;
    const unsigned short* Hb = Hbf + (size_t)b * Sn * Dn;
    const unsigned short* Lb = Lbf  + (size_t)l0 * Dn;   // rows l0..l0+31
    const unsigned short* Wb = Wobf + (size_t)l0 * Dn;

    // Online softmax state per label tile; each lane owns one label column.
    float mx[2]  = {-3.0e38f, -3.0e38f};
    float den[2] = {0.0f, 0.0f};
    float num[2] = {0.0f, 0.0f};

    for (int s0 = 0; s0 < Sn; s0 += 32) {      // 16 passes of 2 s-tiles
        v8f sc[2][2] = {};                     // [s-tile][label-tile]
        v8f hw[2][2] = {};
        for (int k0 = 0; k0 < Dn; k0 += 32) {
            v16bf a0 = load_a_frag(Ab, Dn, s0,      k0, lane);
            v16bf h0 = load_a_frag(Hb, Dn, s0,      k0, lane);
            v16bf a1 = load_a_frag(Ab, Dn, s0 + 16, k0, lane);
            v16bf h1 = load_a_frag(Hb, Dn, s0 + 16, k0, lane);
#pragma unroll
            for (int lj = 0; lj < 2; ++lj) {
                v16bf bl = load_b_frag(Lb, Dn, lj * 16, k0, lane);
                v16bf bw = load_b_frag(Wb, Dn, lj * 16, k0, lane);
                sc[0][lj] = wmma_bf16(a0, bl, sc[0][lj]);
                hw[0][lj] = wmma_bf16(h0, bw, hw[0][lj]);
                sc[1][lj] = wmma_bf16(a1, bl, sc[1][lj]);
                hw[1][lj] = wmma_bf16(h1, bw, hw[1][lj]);
            }
        }
        // Tile-max-first online softmax: one rescale exp per tile, then a
        // single exp per score element (halves v_exp_f32 vs the naive update).
#pragma unroll
        for (int lj = 0; lj < 2; ++lj) {
            float tmax = sc[0][lj][0];
#pragma unroll
            for (int si = 0; si < 2; ++si)
#pragma unroll
                for (int r = 0; r < 8; ++r)
                    tmax = fmaxf(tmax, sc[si][lj][r]);

            float nm   = fmaxf(mx[lj], tmax);
            float corr = __expf(mx[lj] - nm);
            float d    = den[lj] * corr;
            float u    = num[lj] * corr;
#pragma unroll
            for (int si = 0; si < 2; ++si)
#pragma unroll
                for (int r = 0; r < 8; ++r) {
                    float e = __expf(sc[si][lj][r] - nm);
                    d += e;
                    u += e * hw[si][lj][r];
                }
            den[lj] = d;
            num[lj] = u;
            mx[lj]  = nm;
        }
    }

    // Combine the two lanes (lane, lane^16) that share a label column (wave32).
    int n = lane & 15;
#pragma unroll
    for (int lj = 0; lj < 2; ++lj) {
        float mx2  = __shfl_xor(mx[lj],  16, 32);
        float den2 = __shfl_xor(den[lj], 16, 32);
        float num2 = __shfl_xor(num[lj], 16, 32);
        float M  = fmaxf(mx[lj], mx2);
        float c1 = __expf(mx[lj] - M), c2 = __expf(mx2 - M);
        float Dt = den[lj] * c1 + den2 * c2;
        float Nt = num[lj] * c1 + num2 * c2;
        float logit = Nt / Dt + b_out[l0 + lj * 16 + n];
        float prob  = 1.0f / (1.0f + __expf(-logit));
        if (lane < 16) out[(size_t)b * Ln + l0 + lj * 16 + n] = prob;
    }
}

// ---------------------------------------------------------------------------
extern "C" void kernel_launch(void* const* d_in, const int* in_sizes, int n_in,
                              void* d_out, int out_size, void* d_ws, size_t ws_size,
                              hipStream_t stream) {
    const float* H         = (const float*)d_in[0];
    const float* label_emb = (const float*)d_in[1];
    const float* W_att     = (const float*)d_in[2];
    const float* b_att     = (const float*)d_in[3];
    const float* W_out     = (const float*)d_in[4];
    const float* b_out     = (const float*)d_in[5];
    float* out = (float*)d_out;

    const int HSZ = Bn * Sn * Dn;   // 3,145,728
    const int LSZ = Ln * Dn;        // 3,145,728
    const int WSZ = Dn * Dn;        //   589,824

    unsigned short* Hbf   = (unsigned short*)d_ws;     // bf16 H
    unsigned short* Abf   = Hbf  + HSZ;                // bf16 tanh-projected A
    unsigned short* Lbf   = Abf  + HSZ;                // bf16 label_emb
    unsigned short* Wobf  = Lbf  + LSZ;                // bf16 W_out
    unsigned short* WattT = Wobf + LSZ;                // bf16 W_att transposed
    // total: (2*HSZ + 2*LSZ + WSZ) * 2 bytes ~= 26.3 MB of workspace

    cvt_f32_bf16<<<(HSZ + 255) / 256, 256, 0, stream>>>(H, Hbf, HSZ);
    cvt_f32_bf16<<<(LSZ + 255) / 256, 256, 0, stream>>>(label_emb, Lbf, LSZ);
    cvt_f32_bf16<<<(LSZ + 255) / 256, 256, 0, stream>>>(W_out, Wobf, LSZ);
    cvt_transpose<<<(WSZ + 255) / 256, 256, 0, stream>>>(W_att, WattT);

    // GEMM1: (B*S/16)*(D/16) = 256*48 = 12288 wave tiles, 8 waves/block.
    lwan_proj_tanh<<<12288 / 8, 256, 0, stream>>>(Hbf, WattT, b_att, Abf);

    // Fused attention: 8 batches * 128 label groups = 1024 waves, 8/block.
    lwan_attn_logits<<<1024 / 8, 256, 0, stream>>>(Abf, Hbf, Lbf, Wobf, b_out, out);
}
